// Block_15350213115931
// MI455X (gfx1250) — matrix operand order repaired and checked by reading, not compiled, not run on hardware
//
#include <hip/hip_runtime.h>
#include <math.h>

typedef _Float16 h16;
typedef __attribute__((ext_vector_type(16))) _Float16 v16h;
typedef __attribute__((ext_vector_type(8)))  _Float16 v8h;
typedef __attribute__((ext_vector_type(8)))  float    v8f;
typedef __attribute__((ext_vector_type(4)))  float    v4f;
typedef __attribute__((ext_vector_type(4)))  unsigned int v4u;
typedef __attribute__((ext_vector_type(8)))  int      v8i;
typedef __attribute__((ext_vector_type(4)))  int      v4i;

constexpr int Bc  = 2;
constexpr int Tc  = 2048;
constexpr int Ec  = 1024;
constexpr int Hc  = 16;
constexpr int HDc = 64;
constexpr int Mc  = Bc * Tc;   // 4096 rows

#define DEV static __device__ __forceinline__

// ---- WMMA fragment loaders (CDNA5 16x16x32 f16 layouts, wave32) ----
// A (16x32): lane = row m = lane&15; k elements = h*16+e
DEV v16h ld_frag_a(const h16* base, int stride, int lane) {
  int m = lane & 15, hh = (lane >> 4) & 1;
  const h16* r = base + m * stride;
  return *(const v16h*)(r + hh * 16);
}
// B (32x16): lane = col n; k elements = h*16+e; base laid out [n][k] -> contiguous 32B
DEV v16h ld_frag_b(const h16* base, int stride, int lane) {
  int n = lane & 15, hh = (lane >> 4) & 1;
  return *(const v16h*)(base + n * stride + hh * 16);
}
DEV v8f wmma16(v16h a, v16h b, v8f c) {
  return __builtin_amdgcn_wmma_f32_16x16x32_f16(false, a, false, b, (short)0, c,
                                                false, false);
}

// ---- Tensor Data Mover: 2D tile load (f16) Global -> LDS -------------------
// Tile: 128 rows x 32 f16 elements, row-major, global row stride = strideElems.
// D# packing per CDNA5 ISA (2D: groups 2/3 zero).
// Toolchain uses the 6-arg builtin: (v4u, v8i, v4i, v4i, v8i, i32 cpol).
DEV void tdm_load_a_tile(const h16* gsrc, unsigned lds_byte_off,
                         unsigned strideElems) {
  unsigned long long ga = (unsigned long long)(uintptr_t)gsrc;
  v4u g0;
  g0[0] = 1u;                                    // count=1, user descriptor
  g0[1] = lds_byte_off;                          // lds_addr [63:32]
  g0[2] = (unsigned)(ga & 0xffffffffu);          // global_addr lo
  g0[3] = (unsigned)((ga >> 32) & 0x1ffffffu) | (2u << 30);  // addr hi | type=2
  v8i g1;
  g1[0] = 0x00010000;                            // data_size=1 (2 bytes)
  g1[1] = (int)((32u & 0xffffu) << 16);          // tensor_dim0 = 32 (lo16<<16)
  g1[2] = (int)(((32u >> 16) & 0xffffu) | ((128u & 0xffffu) << 16));  // d0 hi|d1 lo
  g1[3] = (int)(((128u >> 16) & 0xffffu) | (32u << 16));  // d1 hi | tile_dim0=32
  g1[4] = (int)128u;                             // tile_dim1=128, tile_dim2=0
  g1[5] = (int)strideElems;                      // tensor_dim0_stride lo32
  g1[6] = 0;                                     // stride hi / dim1_stride lo
  g1[7] = 0;
  v4i z4; z4[0] = 0; z4[1] = 0; z4[2] = 0; z4[3] = 0;
  v8i z8;
#pragma unroll
  for (int i = 0; i < 8; ++i) z8[i] = 0;
  __builtin_amdgcn_tensor_load_to_lds(g0, g1, z4, z4, z8, 0);
}

// ---------------- LayerNorm (f32 in -> f16 out), one block per row ----------
__global__ __launch_bounds__(256) void ln_f16(const float* __restrict__ x,
                                              const float* __restrict__ g,
                                              const float* __restrict__ b,
                                              h16* __restrict__ out) {
  const int row = blockIdx.x;
  const float* xr = x + (size_t)row * Ec;
  float v[4], s = 0.f, s2 = 0.f;
#pragma unroll
  for (int i = 0; i < 4; ++i) {
    v[i] = xr[threadIdx.x + i * 256];
    s += v[i];
    s2 += v[i] * v[i];
  }
#pragma unroll
  for (int off = 16; off > 0; off >>= 1) {
    s  += __shfl_xor(s, off, 32);
    s2 += __shfl_xor(s2, off, 32);
  }
  __shared__ float red[24];
  int lane = threadIdx.x & 31, wid = threadIdx.x >> 5;
  if (lane == 0) { red[wid] = s; red[8 + wid] = s2; }
  __syncthreads();
  if (threadIdx.x == 0) {
    float ts = 0.f, ts2 = 0.f;
    for (int i = 0; i < 8; ++i) { ts += red[i]; ts2 += red[8 + i]; }
    red[16] = ts; red[17] = ts2;
  }
  __syncthreads();
  float mu  = red[16] * (1.0f / Ec);
  float var = red[17] * (1.0f / Ec) - mu * mu;
  float rstd = rsqrtf(var + 1e-5f);
#pragma unroll
  for (int i = 0; i < 4; ++i) {
    int c = threadIdx.x + i * 256;
    float o = (v[i] - mu) * rstd * g[c] + b[c];
    out[(size_t)row * Ec + c] = (h16)o;
  }
}

// ---------------- WMMA GEMM: C[M,N] = A_f16[M,K] @ W_f32->f16[K,N] ----------
// Double-buffered: A tiles via TDM (TENSORcnt), W tiles via vector loads with
// on-the-fly f32->f16 conversion, prefetched into registers during compute.
// EPI 0: +bias, scatter f16 into q/k/v [B,H,T,HD]
// EPI 1: +bias +resid -> f32
// EPI 2: +bias, GELU  -> f16
// EPI 3: +bias +resid -> f32 (final out)
template <int EPI>
__global__ __launch_bounds__(256) void gemm_wmma(
    const h16* __restrict__ A, const float* __restrict__ W,
    const float* __restrict__ bias, const float* __restrict__ resid,
    float* __restrict__ outf, h16* __restrict__ outh, h16* __restrict__ qb,
    h16* __restrict__ kb, h16* __restrict__ vb, int K, int N) {
  __shared__ __attribute__((aligned(32))) h16 sA[2][128 * 32];  // [m][k]
  __shared__ __attribute__((aligned(32))) h16 sB[2][128 * 32];  // [n][k]
  const int tid = threadIdx.x, lane = tid & 31, wid = tid >> 5;
  const int wm = wid >> 2, wn = wid & 3;
  const int m0 = blockIdx.y * 128, n0 = blockIdx.x * 128;

  v8f acc[4][2];
#pragma unroll
  for (int mt = 0; mt < 4; ++mt)
#pragma unroll
    for (int nt = 0; nt < 2; ++nt)
#pragma unroll
      for (int i = 0; i < 8; ++i) acc[mt][nt][i] = 0.0f;

  // per-thread W-tile mapping (32x128 f32, 4 chunks of 4)
  int wkr[4], wnc[4];
#pragma unroll
  for (int i = 0; i < 4; ++i) {
    int c = tid + i * 256;
    wkr[i] = c >> 5;
    wnc[i] = (c & 31) << 2;
  }

  // ---- prologue: stage tile 0 into buffer 0 ----
  if (wid == 0)
    tdm_load_a_tile(A + (size_t)m0 * K,
                    (unsigned)(uintptr_t)&sA[0][0], (unsigned)K);
  v4f wreg[4];
#pragma unroll
  for (int i = 0; i < 4; ++i)
    wreg[i] = *(const v4f*)(W + (size_t)wkr[i] * N + n0 + wnc[i]);
#pragma unroll
  for (int i = 0; i < 4; ++i)
#pragma unroll
    for (int j = 0; j < 4; ++j)
      sB[0][(wnc[i] + j) * 32 + wkr[i]] = (h16)wreg[i][j];

  for (int k0 = 0; k0 < K; k0 += 32) {
    const int cur = (k0 >> 5) & 1, nxt = cur ^ 1;
    if (wid == 0) __builtin_amdgcn_s_wait_tensorcnt(0);
    __syncthreads();  // waits DScnt then barrier: cur buffers ready

    const bool more = (k0 + 32) < K;
    if (more) {
      // kick off next A tile DMA + next W tile loads while we compute
      if (wid == 0)
        tdm_load_a_tile(A + (size_t)m0 * K + (k0 + 32),
                        (unsigned)(uintptr_t)&sA[nxt][0], (unsigned)K);
#pragma unroll
      for (int i = 0; i < 4; ++i)
        wreg[i] =
            *(const v4f*)(W + (size_t)(k0 + 32 + wkr[i]) * N + n0 + wnc[i]);
    }

    v16h af[4], bfr[2];
#pragma unroll
    for (int mt = 0; mt < 4; ++mt)
      af[mt] = ld_frag_a(&sA[cur][(wm * 64 + mt * 16) * 32], 32, lane);
#pragma unroll
    for (int nt = 0; nt < 2; ++nt)
      bfr[nt] = ld_frag_b(&sB[cur][(wn * 32 + nt * 16) * 32], 32, lane);
#pragma unroll
    for (int mt = 0; mt < 4; ++mt)
#pragma unroll
      for (int nt = 0; nt < 2; ++nt)
        acc[mt][nt] = wmma16(af[mt], bfr[nt], acc[mt][nt]);

    if (more) {
#pragma unroll
      for (int i = 0; i < 4; ++i)
#pragma unroll
        for (int j = 0; j < 4; ++j)
          sB[nxt][(wnc[i] + j) * 32 + wkr[i]] = (h16)wreg[i][j];
    }
  }

  const int lh = (lane >> 4) & 1, ln = lane & 15;
#pragma unroll
  for (int nt = 0; nt < 2; ++nt) {
    int n = n0 + wn * 32 + nt * 16 + ln;
    float bv = bias[n];
#pragma unroll
    for (int mt = 0; mt < 4; ++mt) {
#pragma unroll
      for (int r = 0; r < 8; ++r) {
        int m = m0 + wm * 64 + mt * 16 + r + 8 * lh;
        float val = acc[mt][nt][r] + bv;
        if (EPI == 0) {
          int which = n >> 10, e = n & 1023, head = e >> 6, hd = e & 63;
          h16* dst = (which == 0) ? qb : (which == 1) ? kb : vb;
          int bb = m >> 11, t = m & 2047;  // T = 2048
          dst[((size_t)(bb * Hc + head) * Tc + t) * HDc + hd] = (h16)val;
        } else if (EPI == 1 || EPI == 3) {
          size_t idx = (size_t)m * N + n;
          outf[idx] = val + resid[idx];
        } else {  // EPI == 2: GELU (tanh approx) -> f16
          float x3 = val * val * val;
          float ge = 0.5f * val *
                     (1.0f + tanhf(0.7978845608028654f * (val + 0.044715f * x3)));
          outh[(size_t)m * N + n] = (h16)ge;
        }
      }
    }
  }
}

// ---------------- Flash attention (causal), WMMA f16 ------------------------
// grid: (T/64, B*H), block: 128 threads = 4 waves; wave w owns q rows
// [qtile*64 + w*16, +16). Online softmax in C-fragment layout.
__global__ __launch_bounds__(128) void attn_wmma(const h16* __restrict__ q,
                                                 const h16* __restrict__ k,
                                                 const h16* __restrict__ v,
                                                 h16* __restrict__ y) {
  __shared__ __attribute__((aligned(32))) h16 sV[64 * 32];     // [hd][key]
  __shared__ __attribute__((aligned(32))) h16 sP[4][16 * 32];  // per-wave P
  const int qtile = blockIdx.x, bh = blockIdx.y;
  const int b = bh >> 4, hh = bh & 15;  // H = 16
  const int tid = threadIdx.x, lane = tid & 31, wid = tid >> 5;
  const int lh = (lane >> 4) & 1, ln = lane & 15;
  const h16* qh = q + (size_t)bh * Tc * HDc;
  const h16* kh = k + (size_t)bh * Tc * HDc;
  const h16* vh = v + (size_t)bh * Tc * HDc;
  const int qr0 = qtile * 64 + wid * 16;

  v16h aq0 = ld_frag_a(qh + (size_t)qr0 * HDc, HDc, lane);
  v16h aq1 = ld_frag_a(qh + (size_t)qr0 * HDc + 32, HDc, lane);

  v8f o[4];
  float mrow[8], lrow[8];
#pragma unroll
  for (int j = 0; j < 4; ++j)
#pragma unroll
    for (int r = 0; r < 8; ++r) o[j][r] = 0.0f;
#pragma unroll
  for (int r = 0; r < 8; ++r) { mrow[r] = -3.0e38f; lrow[r] = 0.0f; }

  const float scale = 0.125f;  // 1/sqrt(64)
  const int kv_end = qtile * 64 + 64;

  for (int kv0 = 0; kv0 < kv_end; kv0 += 32) {
    __syncthreads();
#pragma unroll
    for (int i = 0; i < 2; ++i) {  // stage V tile transposed: [hd][key]
      int c = tid + i * 128;
      int key = c >> 3, hd0 = (c & 7) << 3;
      v8h vv = *(const v8h*)(vh + (size_t)(kv0 + key) * HDc + hd0);
#pragma unroll
      for (int j = 0; j < 8; ++j) sV[(hd0 + j) * 32 + key] = vv[j];
    }
    __syncthreads();

    // S = Q @ K^T : 16q x 32keys (two 16x16 C tiles), K-frags straight from HBM
    v8f s0, s1;
#pragma unroll
    for (int i = 0; i < 8; ++i) { s0[i] = 0.0f; s1[i] = 0.0f; }
    s0 = wmma16(aq0, ld_frag_b(kh + (size_t)kv0 * HDc, HDc, lane), s0);
    s0 = wmma16(aq1, ld_frag_b(kh + (size_t)kv0 * HDc + 32, HDc, lane), s0);
    s1 = wmma16(aq0, ld_frag_b(kh + (size_t)(kv0 + 16) * HDc, HDc, lane), s1);
    s1 = wmma16(aq1, ld_frag_b(kh + (size_t)(kv0 + 16) * HDc + 32, HDc, lane), s1);

    float fac[8];
#pragma unroll
    for (int r = 0; r < 8; ++r) {
      int qrow = qr0 + r + 8 * lh;
      float a0 = s0[r] * scale;
      float a1 = s1[r] * scale;
      if (kv0 + ln > qrow) a0 = -3.0e38f;       // causal mask
      if (kv0 + 16 + ln > qrow) a1 = -3.0e38f;
      float rm = fmaxf(a0, a1);
#pragma unroll
      for (int off = 1; off < 16; off <<= 1) rm = fmaxf(rm, __shfl_xor(rm, off, 32));
      float mnew = fmaxf(mrow[r], rm);
      float f = __expf(mrow[r] - mnew);
      float p0 = __expf(a0 - mnew);
      float p1 = __expf(a1 - mnew);
      float rs = p0 + p1;
#pragma unroll
      for (int off = 1; off < 16; off <<= 1) rs += __shfl_xor(rs, off, 32);
      lrow[r] = lrow[r] * f + rs;
      mrow[r] = mnew;
      fac[r] = f;
      s0[r] = p0;
      s1[r] = p1;
    }
#pragma unroll
    for (int j = 0; j < 4; ++j)
#pragma unroll
      for (int r = 0; r < 8; ++r) o[j][r] *= fac[r];

    // transpose P (C-layout) -> A-layout via per-wave LDS scratch
    h16* lp = &sP[wid][0];
#pragma unroll
    for (int r = 0; r < 8; ++r) {
      int row = r + 8 * lh;
      lp[row * 32 + ln] = (h16)s0[r];
      lp[row * 32 + 16 + ln] = (h16)s1[r];
    }
    v16h pf = ld_frag_a(lp, 32, lane);  // same-wave DS ordering
#pragma unroll
    for (int j = 0; j < 4; ++j)
      o[j] = wmma16(pf, ld_frag_b(sV + j * 16 * 32, 32, lane), o[j]);
  }

#pragma unroll
  for (int r = 0; r < 8; ++r) {
    int t = qr0 + r + 8 * lh;
    float inv = 1.0f / lrow[r];
#pragma unroll
    for (int j = 0; j < 4; ++j)
      y[((size_t)(b * Tc + t)) * Ec + hh * HDc + j * 16 + ln] =
          (h16)(o[j][r] * inv);
  }
}

// ---------------- Host launcher ---------------------------------------------
extern "C" void kernel_launch(void* const* d_in, const int* in_sizes, int n_in,
                              void* d_out, int out_size, void* d_ws,
                              size_t ws_size, hipStream_t stream) {
  (void)in_sizes; (void)n_in; (void)out_size; (void)ws_size;
  const float* x      = (const float*)d_in[0];
  const float* ln1_g  = (const float*)d_in[1];
  const float* ln1_b  = (const float*)d_in[2];
  const float* w_attn = (const float*)d_in[3];
  const float* b_attn = (const float*)d_in[4];
  const float* w_proj = (const float*)d_in[5];
  const float* b_proj = (const float*)d_in[6];
  const float* ln2_g  = (const float*)d_in[7];
  const float* ln2_b  = (const float*)d_in[8];
  const float* w_fc   = (const float*)d_in[9];
  const float* b_fc   = (const float*)d_in[10];
  const float* w_fc2  = (const float*)d_in[11];
  const float* b_fc2  = (const float*)d_in[12];
  float* out = (float*)d_out;

  char* p = (char*)d_ws;
  auto carve = [&](size_t bytes) {
    char* r = p;
    p += (bytes + 255) & ~(size_t)255;
    return r;
  };
  const size_t ME = (size_t)Mc * Ec;
  h16*   hln = (h16*)carve(ME * 2);                   // LN1 out
  h16*   qb  = (h16*)carve(ME * 2);                   // q [B,H,T,HD]
  h16*   kb  = (h16*)carve(ME * 2);                   // k
  h16*   vb  = (h16*)carve(ME * 2);                   // v
  h16*   yb  = (h16*)carve(ME * 2);                   // attn out [B,T,E]
  float* x1  = (float*)carve(ME * 4);                 // x + proj (f32)
  h16*   h2  = (h16*)carve(ME * 2);                   // LN2 out
  h16*   h1  = (h16*)carve((size_t)Mc * 4 * Ec * 2);  // gelu(fc1) [M,4E]

  ln_f16<<<Mc, 256, 0, stream>>>(x, ln1_g, ln1_b, hln);

  gemm_wmma<0><<<dim3(3 * Ec / 128, Mc / 128), 256, 0, stream>>>(
      hln, w_attn, b_attn, nullptr, nullptr, nullptr, qb, kb, vb, Ec, 3 * Ec);

  attn_wmma<<<dim3(Tc / 64, Bc * Hc), 128, 0, stream>>>(qb, kb, vb, yb);

  gemm_wmma<1><<<dim3(Ec / 128, Mc / 128), 256, 0, stream>>>(
      yb, w_proj, b_proj, x, x1, nullptr, nullptr, nullptr, nullptr, Ec, Ec);

  ln_f16<<<Mc, 256, 0, stream>>>(x1, ln2_g, ln2_b, h2);

  gemm_wmma<2><<<dim3(4 * Ec / 128, Mc / 128), 256, 0, stream>>>(
      h2, w_fc, b_fc, nullptr, nullptr, h1, nullptr, nullptr, nullptr, Ec,
      4 * Ec);

  gemm_wmma<3><<<dim3(Ec / 128, Mc / 128), 256, 0, stream>>>(
      h1, w_fc2, b_fc2, x1, out, nullptr, nullptr, nullptr, nullptr, 4 * Ec,
      Ec);
}
